// GraphAttentionLayer_89172110999821
// MI455X (gfx1250) — matrix operand order repaired and checked by reading, not compiled, not run on hardware
//
#include <hip/hip_runtime.h>
#include <hip/hip_bf16.h>

typedef _Float16 f16;
typedef __attribute__((ext_vector_type(16))) _Float16 v16h;
typedef __attribute__((ext_vector_type(8)))  _Float16 v8h;
typedef __attribute__((ext_vector_type(4)))  _Float16 v4h;
typedef __attribute__((ext_vector_type(8)))  float    v8f;

#define N_NODES 8192
#define FDIM    256
#define GAT_ALPHA 0.2f
#define NEG_BIG (-1.0e12f)

static __device__ __forceinline__ v8f zero8() {
  v8f z;
#pragma unroll
  for (int i = 0; i < 8; ++i) z[i] = 0.f;
  return z;
}

// Load a 16x32 f16 A-fragment from LDS in contiguous k order (same order as the B loads):
// lane m = lane&15 holds row m; half = lane>>4;
// elements 0..15 -> K = kbase + 16*half + p
static __device__ __forceinline__ v16h lds_frag_a(const f16* base, int row, int pitch,
                                                  int half, int kbase) {
  const f16* p = base + row * pitch + kbase + 16 * half;
  v8h lo = *(const v8h*)p;
  v8h hi = *(const v8h*)(p + 8);
  v16h r;
#pragma unroll
  for (int i = 0; i < 8; ++i) { r[i] = lo[i]; r[i + 8] = hi[i]; }
  return r;
}

// ---------------- kernel 1: u = W @ a[:F], v = W @ a[F:]  (tiny) ----------------
__global__ void k_uv(const float* __restrict__ W, const float* __restrict__ a,
                     float* __restrict__ u, float* __restrict__ v) {
  int k = threadIdx.x;  // 0..255
  float su = 0.f, sv = 0.f;
  for (int f = 0; f < FDIM; ++f) {
    float w = W[k * FDIM + f];
    su += w * a[f];
    sv += w * a[FDIM + f];
  }
  u[k] = su;
  v[k] = sv;
}

// ---------------- transpose f32 [R][C] -> f16 [C][R] ----------------
__global__ void k_transpose_h(const float* __restrict__ in, f16* __restrict__ out,
                              int R, int C) {
  __shared__ float tile[32][33];
  int x = blockIdx.x * 32 + threadIdx.x;   // input col
  int y = blockIdx.y * 32 + threadIdx.y;   // input row
#pragma unroll
  for (int r = 0; r < 32; r += 8)
    tile[threadIdx.y + r][threadIdx.x] = in[(size_t)(y + r) * C + x];
  __syncthreads();
  int x2 = blockIdx.y * 32 + threadIdx.x;  // output col (= input row)
  int y2 = blockIdx.x * 32 + threadIdx.y;  // output row (= input col)
#pragma unroll
  for (int r = 0; r < 32; r += 8)
    out[(size_t)(y2 + r) * R + x2] = (f16)tile[threadIdx.x][threadIdx.y + r];
}

// ---------------- kernel 2: s1[i]=inp[i].u, s2[i]=inp[i].v  (wave per row) ------
__global__ void k_scores(const float* __restrict__ inp, const float* __restrict__ u,
                         const float* __restrict__ v, float* __restrict__ s1,
                         float* __restrict__ s2) {
  int lane = threadIdx.x & 31;
  int row = blockIdx.x * 8 + (threadIdx.x >> 5);
  float a1 = 0.f, a2 = 0.f;
#pragma unroll
  for (int t = 0; t < FDIM / 32; ++t) {
    int c = t * 32 + lane;
    float x = inp[(size_t)row * FDIM + c];
    a1 += x * u[c];
    a2 += x * v[c];
  }
#pragma unroll
  for (int off = 16; off >= 1; off >>= 1) {
    a1 += __shfl_xor(a1, off);
    a2 += __shfl_xor(a2, off);
  }
  if (lane == 0) { s1[row] = a1; s2[row] = a2; }
}

// ---------------- kernel 3: online softmax stats per row (pos & neg) ------------
__global__ void k_stats(const int* __restrict__ adj, const float* __restrict__ s1,
                        const float* __restrict__ s2, float* __restrict__ mp,
                        float* __restrict__ lp, float* __restrict__ mn,
                        float* __restrict__ ln) {
  int lane = threadIdx.x & 31;
  int row = blockIdx.x * 8 + (threadIdx.x >> 5);
  float s1i = s1[row];
  const int* arow = adj + (size_t)row * N_NODES;
  float m1 = -INFINITY, l1 = 0.f, m2 = -INFINITY, l2 = 0.f;
  for (int jc = 0; jc < N_NODES; jc += 32) {
    int j = jc + lane;
    int a = arow[j];
    float t = s1i + s2[j];
    float xb = t > 0.f ? t : GAT_ALPHA * t;
    float x  = a ? xb  : NEG_BIG;   // masked entries use -1e12 like the reference
    float x2 = a ? -xb : NEG_BIG;
    if (x > m1) { l1 = l1 * __expf(m1 - x) + 1.f; m1 = x; } else l1 += __expf(x - m1);
    if (x2 > m2) { l2 = l2 * __expf(m2 - x2) + 1.f; m2 = x2; } else l2 += __expf(x2 - m2);
  }
#pragma unroll
  for (int off = 16; off >= 1; off >>= 1) {
    float mo = __shfl_xor(m1, off), lo = __shfl_xor(l1, off);
    float M = fmaxf(m1, mo);
    l1 = l1 * __expf(m1 - M) + lo * __expf(mo - M); m1 = M;
    mo = __shfl_xor(m2, off); lo = __shfl_xor(l2, off);
    M = fmaxf(m2, mo);
    l2 = l2 * __expf(m2 - M) + lo * __expf(mo - M); m2 = M;
  }
  if (lane == 0) { mp[row] = m1; lp[row] = l1; mn[row] = m2; ln[row] = l2; }
}

// ---------------- main kernel: attention GEMMs + fused wtrans GEMM + ELU --------
// Block = 256 threads = 8 waves; tile = 32 rows x 256 cols; K loop over j in 32s.
__global__ __launch_bounds__(256) void k_gat_main(
    const int* __restrict__ adj, const float* __restrict__ s2g,
    const float* __restrict__ s1g, const float* __restrict__ mpg,
    const float* __restrict__ lpg, const float* __restrict__ mng,
    const float* __restrict__ lng, const f16* __restrict__ inp_hT,
    const f16* __restrict__ wt_hT, float* __restrict__ out) {
  __shared__ __align__(16) f16 Apos[32 * 40];   // pitch 40 f16: conflict-free b128 reads
  __shared__ __align__(16) f16 Aneg[32 * 40];
  __shared__ __align__(16) f16 Aout[32 * 520];  // 32 x 512 normalized [hp | hpn], pitch 520
  __shared__ float s1l[32], mpl[32], mnl[32], rlpl[32], rlnl[32];

  const int tid = threadIdx.x;
  const int lane = tid & 31, wid = tid >> 5;
  const int half = lane >> 4, n16 = lane & 15;
  const int rowbase = blockIdx.x * 32;

  if (tid < 32) {
    s1l[tid] = s1g[rowbase + tid];
    mpl[tid] = mpg[rowbase + tid];
    mnl[tid] = mng[rowbase + tid];
    // precompute reciprocals (hw rcp, fine for softmax denominators >= 1),
    // fold the negative-attention sign into rlnl.
    rlpl[tid] = __builtin_amdgcn_rcpf(lpg[rowbase + tid]);
    rlnl[tid] = -__builtin_amdgcn_rcpf(lng[rowbase + tid]);
  }
  __syncthreads();

  v8f Cp[2][2], Cn[2][2];  // [m-tile][n-tile] f32 accumulators
#pragma unroll
  for (int mt = 0; mt < 2; ++mt)
#pragma unroll
    for (int nt = 0; nt < 2; ++nt) { Cp[mt][nt] = zero8(); Cn[mt][nt] = zero8(); }

  // Each thread builds 4 consecutive weights of one A row per chunk.
  const int wm = tid >> 3;        // 0..31 (row within tile)
  const int wk = (tid & 7) * 4;   // 0..28 (k offset within chunk)
  const int* adjrow = adj + (size_t)(rowbase + wm) * N_NODES + wk;
  const float s1m = s1l[wm], mpm = mpl[wm], mnm = mnl[wm];

  const f16* bptr0 = inp_hT + (size_t)(wid * 32 + n16) * N_NODES + 16 * half;
  const f16* bptr1 = inp_hT + (size_t)(wid * 32 + 16 + n16) * N_NODES + 16 * half;

  for (int jb = 0; jb < N_NODES; jb += 32) {
    int4 a4 = *(const int4*)(adjrow + jb);
    float4 s4 = *(const float4*)(s2g + jb + wk);
    if (jb + 32 < N_NODES) __builtin_prefetch(adjrow + jb + 32, 0, 3);

    float sv[4] = {s4.x, s4.y, s4.z, s4.w};
    int av[4] = {a4.x, a4.y, a4.z, a4.w};
    v4h wpv, wnv;
#pragma unroll
    for (int i = 0; i < 4; ++i) {
      float t = s1m + sv[i];
      float xb = t > 0.f ? t : GAT_ALPHA * t;
      float xp = av[i] ? xb : NEG_BIG;
      float xn = av[i] ? -xb : NEG_BIG;
      wpv[i] = (f16)__expf(xp - mpm);   // unnormalized softmax weight, <= 1
      wnv[i] = (f16)__expf(xn - mnm);
    }
    *(v4h*)&Apos[wm * 40 + wk] = wpv;
    *(v4h*)&Aneg[wm * 40 + wk] = wnv;
    __syncthreads();

    // B fragments: contiguous 32B per lane from transposed f16 inp (L2-resident)
    v16h B0 = *(const v16h*)(bptr0 + jb);
    v16h B1 = *(const v16h*)(bptr1 + jb);
#pragma unroll
    for (int mt = 0; mt < 2; ++mt) {
      v16h Ap = lds_frag_a(Apos, mt * 16 + n16, 40, half, 0);
      v16h An = lds_frag_a(Aneg, mt * 16 + n16, 40, half, 0);
      Cp[mt][0] = __builtin_amdgcn_wmma_f32_16x16x32_f16(false, Ap, false, B0, (short)0, Cp[mt][0], false, false);
      Cp[mt][1] = __builtin_amdgcn_wmma_f32_16x16x32_f16(false, Ap, false, B1, (short)0, Cp[mt][1], false, false);
      Cn[mt][0] = __builtin_amdgcn_wmma_f32_16x16x32_f16(false, An, false, B0, (short)0, Cn[mt][0], false, false);
      Cn[mt][1] = __builtin_amdgcn_wmma_f32_16x16x32_f16(false, An, false, B1, (short)0, Cn[mt][1], false, false);
    }
    __syncthreads();
  }

  // Epilogue: normalize, pack [h_prime | -h_prime_neg] into LDS as f16 A'
#pragma unroll
  for (int mt = 0; mt < 2; ++mt)
#pragma unroll
    for (int nt = 0; nt < 2; ++nt)
#pragma unroll
      for (int r = 0; r < 8; ++r) {
        int row = mt * 16 + half * 8 + r;  // C/D layout: VGPR r -> M = 8*half + r
        int col = wid * 32 + nt * 16 + n16;
        Aout[row * 520 + col]       = (f16)(Cp[mt][nt][r] * rlpl[row]);
        Aout[row * 520 + 256 + col] = (f16)(Cn[mt][nt][r] * rlnl[row]);
      }
  __syncthreads();

  // out_tile[32x256] = A'[32x512] @ wtrans[512x256]  (wt_hT is [n][k] f16)
  v8f Co[2][2];
#pragma unroll
  for (int mt = 0; mt < 2; ++mt)
#pragma unroll
    for (int nt = 0; nt < 2; ++nt) Co[mt][nt] = zero8();

  const f16* wptr0 = wt_hT + (size_t)(wid * 32 + n16) * 512 + 16 * half;
  const f16* wptr1 = wt_hT + (size_t)(wid * 32 + 16 + n16) * 512 + 16 * half;
#pragma unroll
  for (int kc = 0; kc < 16; ++kc) {
    int kb = kc * 32;
    v16h B0 = *(const v16h*)(wptr0 + kb);
    v16h B1 = *(const v16h*)(wptr1 + kb);
#pragma unroll
    for (int mt = 0; mt < 2; ++mt) {
      v16h Af = lds_frag_a(Aout, mt * 16 + n16, 520, half, kb);
      Co[mt][0] = __builtin_amdgcn_wmma_f32_16x16x32_f16(false, Af, false, B0, (short)0, Co[mt][0], false, false);
      Co[mt][1] = __builtin_amdgcn_wmma_f32_16x16x32_f16(false, Af, false, B1, (short)0, Co[mt][1], false, false);
    }
  }

  // ELU (alpha=1) + store
#pragma unroll
  for (int mt = 0; mt < 2; ++mt)
#pragma unroll
    for (int nt = 0; nt < 2; ++nt)
#pragma unroll
      for (int r = 0; r < 8; ++r) {
        int grow = rowbase + mt * 16 + half * 8 + r;
        int gcol = wid * 32 + nt * 16 + n16;
        float x = Co[mt][nt][r];
        out[(size_t)grow * FDIM + gcol] = x > 0.f ? x : (__expf(x) - 1.f);
      }
}

extern "C" void kernel_launch(void* const* d_in, const int* in_sizes, int n_in,
                              void* d_out, int out_size, void* d_ws, size_t ws_size,
                              hipStream_t stream) {
  const float* inp    = (const float*)d_in[0];  // [8192,256]
  const int*   adj    = (const int*)d_in[1];    // [8192,8192]
  const float* W      = (const float*)d_in[2];  // [256,256]
  const float* a      = (const float*)d_in[3];  // [512,1]
  const float* wtrans = (const float*)d_in[4];  // [512,256]
  float* out = (float*)d_out;

  // workspace layout (f32 region then f16 region, all >=32B aligned)
  float* u  = (float*)d_ws;          // 256
  float* v  = u + 256;               // 256
  float* s1 = v + 256;               // 8192
  float* s2 = s1 + N_NODES;          // 8192
  float* mp = s2 + N_NODES;          // 8192
  float* lp = mp + N_NODES;          // 8192
  float* mn = lp + N_NODES;          // 8192
  float* ln = mn + N_NODES;          // 8192
  f16* inp_hT = (f16*)(ln + N_NODES);             // [256][8192] f16, 4 MB
  f16* wt_hT  = inp_hT + (size_t)FDIM * N_NODES;  // [256][512]  f16, 256 KB

  k_uv<<<1, 256, 0, stream>>>(W, a, u, v);
  k_transpose_h<<<dim3(FDIM / 32, N_NODES / 32), dim3(32, 8), 0, stream>>>(inp, inp_hT, N_NODES, FDIM);
  k_transpose_h<<<dim3(FDIM / 32, 512 / 32), dim3(32, 8), 0, stream>>>(wtrans, wt_hT, 512, FDIM);
  k_scores<<<N_NODES / 8, 256, 0, stream>>>(inp, u, v, s1, s2);
  k_stats<<<N_NODES / 8, 256, 0, stream>>>(adj, s1, s2, mp, lp, mn, ln);
  k_gat_main<<<N_NODES / 32, 256, 0, stream>>>(adj, s2, s1, mp, lp, mn, ln, inp_hT, wt_hT, out);
}
